// CustomLinearB_49357764166318
// MI455X (gfx1250) — 8-bit matrix operand order repaired and checked by reading, compiled, not run on hardware
//
#include <hip/hip_runtime.h>
#include <cstdint>
#include <cstddef>

typedef int   v8i __attribute__((ext_vector_type(8)));
typedef float v4f __attribute__((ext_vector_type(4)));

#define QMAXF 127.0f

__global__ __launch_bounds__(256) void absmax_x_kernel(const v4f* __restrict__ x4,
                                                       long long n4,
                                                       unsigned int* __restrict__ absmaxBits) {
    float m = 0.0f;
    long long idx = (long long)blockIdx.x * blockDim.x + threadIdx.x;
    long long stride = (long long)gridDim.x * blockDim.x;
    for (long long i = idx; i < n4; i += stride) {
        v4f v = __builtin_nontemporal_load(&x4[i]);
        m = fmaxf(m, fmaxf(fmaxf(fabsf(v.x), fabsf(v.y)), fmaxf(fabsf(v.z), fabsf(v.w))));
    }
    #pragma unroll
    for (int off = 16; off >= 1; off >>= 1)
        m = fmaxf(m, __shfl_xor(m, off, 32));
    __shared__ float smax[8];
    int lane = threadIdx.x & 31;
    int wv = threadIdx.x >> 5;
    if (lane == 0) smax[wv] = m;
    __syncthreads();
    if (threadIdx.x == 0) {
        float b = smax[0];
        #pragma unroll
        for (int i = 1; i < 8; ++i) b = fmaxf(b, smax[i]);
        atomicMax(absmaxBits, __float_as_uint(b));
    }
}

__global__ __launch_bounds__(256) void wquant_kernel(const float* __restrict__ w,
                                                     float* __restrict__ scale_w,
                                                     char* __restrict__ qw,
                                                     int K) {
    int row = blockIdx.x;
    const float4* wr = (const float4*)(w + (size_t)row * K);
    int nv = K >> 2;
    float m = 0.0f;
    for (int i = threadIdx.x; i < nv; i += 256) {
        float4 v = wr[i];
        m = fmaxf(m, fmaxf(fmaxf(fabsf(v.x), fabsf(v.y)), fmaxf(fabsf(v.z), fabsf(v.w))));
    }
    #pragma unroll
    for (int off = 16; off >= 1; off >>= 1)
        m = fmaxf(m, __shfl_xor(m, off, 32));
    __shared__ float smax[8];
    __shared__ float sscale;
    int lane = threadIdx.x & 31;
    int wv = threadIdx.x >> 5;
    if (lane == 0) smax[wv] = m;
    __syncthreads();
    if (threadIdx.x == 0) {
        float b = smax[0];
        #pragma unroll
        for (int i = 1; i < 8; ++i) b = fmaxf(b, smax[i]);
        float s = fmaxf(b, 1e-8f) / QMAXF;
        scale_w[row] = s;
        sscale = s;
    }
    __syncthreads();
    float inv = 1.0f / sscale;
    char4* qr = (char4*)(qw + (size_t)row * K);
    for (int i = threadIdx.x; i < nv; i += 256) {
        float4 v = wr[i];
        char4 q;
        q.x = (char)(int)rintf(fminf(fmaxf(v.x * inv, -QMAXF), QMAXF));
        q.y = (char)(int)rintf(fminf(fmaxf(v.y * inv, -QMAXF), QMAXF));
        q.z = (char)(int)rintf(fminf(fmaxf(v.z * inv, -QMAXF), QMAXF));
        q.w = (char)(int)rintf(fminf(fmaxf(v.w * inv, -QMAXF), QMAXF));
        qr[i] = q;
    }
}

__global__ __launch_bounds__(256) void xquant_kernel(const v4f* __restrict__ x4,
                                                     long long n4,
                                                     const unsigned int* __restrict__ absmaxBits,
                                                     char4* __restrict__ qx4) {
    float sx = fmaxf(__uint_as_float(*absmaxBits), 1e-8f) / QMAXF;
    float inv = 1.0f / sx;
    long long idx = (long long)blockIdx.x * blockDim.x + threadIdx.x;
    long long stride = (long long)gridDim.x * blockDim.x;
    for (long long i = idx; i < n4; i += stride) {
        v4f v = __builtin_nontemporal_load(&x4[i]);
        char4 q;
        q.x = (char)(int)rintf(fminf(fmaxf(v.x * inv, -QMAXF), QMAXF));
        q.y = (char)(int)rintf(fminf(fmaxf(v.y * inv, -QMAXF), QMAXF));
        q.z = (char)(int)rintf(fminf(fmaxf(v.z * inv, -QMAXF), QMAXF));
        q.w = (char)(int)rintf(fminf(fmaxf(v.w * inv, -QMAXF), QMAXF));
        qx4[i] = q;
    }
}

__global__ __launch_bounds__(256) void gemm_i8_wmma_kernel(const char* __restrict__ qx,
                                                           const char* __restrict__ qw,
                                                           const float* __restrict__ scale_w,
                                                           const unsigned int* __restrict__ absmaxBits,
                                                           const float* __restrict__ bias,
                                                           float* __restrict__ out,
                                                           int Ndim, int K) {
    const int lane  = threadIdx.x & 31;
    const int wv    = threadIdx.x >> 5;
    const int waveM = wv & 3;
    const int waveN = wv >> 2;
    const int tileM = blockIdx.y * 128 + waveM * 32;
    const int tileN = blockIdx.x * 128 + waveN * 64;
    const int r16   = lane & 15;
    const int hi    = lane >> 4;

    const char* aBase = qx + (size_t)(tileM + r16) * K + (hi << 4);
    const char* bBase = qw + (size_t)(tileN + r16) * K + (hi << 4);

    v8i zero = {0, 0, 0, 0, 0, 0, 0, 0};
    v8i acc[2][4];
    #pragma unroll
    for (int tm = 0; tm < 2; ++tm)
        #pragma unroll
        for (int tn = 0; tn < 4; ++tn)
            acc[tm][tn] = zero;

    for (int k = 0; k < K; k += 64) {
        v8i A[2];
        #pragma unroll
        for (int tm = 0; tm < 2; ++tm) {
            const int2* p = reinterpret_cast<const int2*>(aBase + (size_t)tm * 16 * K + k);
            int2 q0 = p[0];
            int2 q1 = p[1];
            int2 q2 = p[4];
            int2 q3 = p[5];
            v8i a;
            a[0] = q0.x; a[1] = q0.y;
            a[2] = q1.x; a[3] = q1.y;
            a[4] = q2.x; a[5] = q2.y;
            a[6] = q3.x; a[7] = q3.y;
            A[tm] = a;
        }
        #pragma unroll
        for (int tn = 0; tn < 4; ++tn) {
            const int4* p = reinterpret_cast<const int4*>(bBase + (size_t)tn * 16 * K + k);
            int4 r0 = p[0];
            int4 r1 = p[2];
            v8i b;
            b[0] = r0.x; b[1] = r0.y; b[2] = r0.z; b[3] = r0.w;
            b[4] = r1.x; b[5] = r1.y; b[6] = r1.z; b[7] = r1.w;
            #pragma unroll
            for (int tm = 0; tm < 2; ++tm)
                acc[tm][tn] = __builtin_amdgcn_wmma_i32_16x16x64_iu8(
                     true, A[tm],  true, b,
                    acc[tm][tn],  false,  false);
        }
    }

    const float sx = fmaxf(__uint_as_float(*absmaxBits), 1e-8f) / QMAXF;
    #pragma unroll
    for (int tn = 0; tn < 4; ++tn) {
        const int n  = tileN + tn * 16 + r16;
        const float s  = sx * scale_w[n];
        const float bb = bias[n];
        #pragma unroll
        for (int tm = 0; tm < 2; ++tm) {
            const int mbase = tileM + tm * 16 + hi * 8;
            #pragma unroll
            for (int r = 0; r < 8; ++r) {
                float val = (float)acc[tm][tn][r] * s + bb;
                __builtin_nontemporal_store(val, &out[(size_t)(mbase + r) * Ndim + n]);
            }
        }
    }
}

extern "C" void kernel_launch(void* const* d_in, const int* in_sizes, int n_in,
                              void* d_out, int out_size, void* d_ws, size_t ws_size,
                              hipStream_t stream) {
    const float* x    = (const float*)d_in[0];
    const float* w    = (const float*)d_in[1];
    const float* bias = (const float*)d_in[2];
    float* out = (float*)d_out;

    const int Dout = in_sizes[2];
    const int Din  = in_sizes[1] / Dout;
    const long long nx = (long long)in_sizes[0];
    const int M = (int)(nx / Din);

    char* ws = (char*)d_ws;
    unsigned int* absmaxBits = (unsigned int*)ws;
    float* scale_w = (float*)(ws + 256);
    char*  qw = ws + 256 + (size_t)Dout * sizeof(float);
    char*  qx = qw + (size_t)Dout * Din;

    (void)hipMemsetAsync(absmaxBits, 0, sizeof(unsigned int), stream);

    absmax_x_kernel<<<2048, 256, 0, stream>>>((const v4f*)x, nx >> 2, absmaxBits);
    wquant_kernel<<<Dout, 256, 0, stream>>>(w, scale_w, qw, Din);
    xquant_kernel<<<8192, 256, 0, stream>>>((const v4f*)x, nx >> 2, absmaxBits, (char4*)qx);

    dim3 grid(Dout / 128, M / 128);
    gemm_i8_wmma_kernel<<<grid, 256, 0, stream>>>(qx, qw, scale_w, absmaxBits, bias, out, Dout, Din);
}
